// DecNFM_rate_61203283968772
// MI455X (gfx1250) — matrix operand order repaired and checked by reading, not compiled, not run on hardware
//
#include <hip/hip_runtime.h>
#include <stdint.h>

// ---------------------------------------------------------------------------
// DecNFM rate loss on MI455X (gfx1250, wave32).
// fp32 everywhere except the two 16384x512x512 GEMMs, which use
// v_wmma_f32_16x16x32_bf16 (bf16 inputs, f32 accumulate).
// GEMM: wave tile 32x64 (8 wmma / K-step), register double-buffered K pipeline.
// ---------------------------------------------------------------------------

typedef __bf16 bf16;
typedef __attribute__((ext_vector_type(16))) __bf16 v16bf;
typedef __attribute__((ext_vector_type(8)))  float  v8f;

#define BATCH   16384
#define DMODEL  512
#define NCATE   2000

union Frag16 { uint4 q[2]; v16bf v; bf16 e[16]; };

// ---------------------------------------------------------------- init
__global__ void k_zero_init(float* cs, float* css, float* reg_acc, float* mse_acc) {
  int i = threadIdx.x;
  if (i < DMODEL) { cs[i] = 0.f; css[i] = 0.f; }
  if (i == 0) { *reg_acc = 0.f; *mse_acc = 0.f; }
}

// ------------------------------------------------- batch-independent cate stats
__global__ void k_cate_stats(const float* __restrict__ cate_w,
                             const float* __restrict__ prior,
                             float* __restrict__ cs, float* __restrict__ css) {
  int d  = blockIdx.x * 256 + threadIdx.x;     // 0..511
  int r0 = blockIdx.y * 200;                   // 10 chunks of 200 rows
  float scs = 0.f, scss = 0.f;
  for (int r = r0; r < r0 + 200; ++r) {
    float wc = cate_w[(size_t)r * DMODEL + d] * prior[r];
    scs  += wc;
    scss += wc * wc;
  }
  atomicAdd(&cs[d],  scs);
  atomicAdd(&css[d], scss);
}

// ----------------------------------------- weight convert+transpose f32->bf16
// Output layout [N][K] so each lane's B fragment is two 16-byte contiguous runs along K.
__global__ void k_convert_wT(const float* __restrict__ w, bf16* __restrict__ wT) {
  int c = blockIdx.x * 16 + threadIdx.x;  // col (N)
  int r = blockIdx.y * 16 + threadIdx.y;  // row (K)
  wT[(size_t)c * DMODEL + r] = (bf16)w[(size_t)r * DMODEL + c];
}

// -------------------------------------------- gather + double FM + reg sum
__global__ __launch_bounds__(128)
void k_fm(const int* __restrict__ user, const int* __restrict__ item,
          const int* __restrict__ cate,
          const float* __restrict__ user_w, const float* __restrict__ item_w,
          const float* __restrict__ cate_w,
          const float* __restrict__ cs, const float* __restrict__ css,
          bf16* __restrict__ fm_out, float* __restrict__ reg_acc) {
  __shared__ float red[128];
  int b = blockIdx.x;
  int t = threadIdx.x;
  const float* ur = user_w + (size_t)user[b] * DMODEL;
  const float* ir = item_w + (size_t)item[b] * DMODEL;
  const float* cr = cate_w + (size_t)cate[b] * DMODEL;
  float regl = 0.f;
  #pragma unroll
  for (int i = 0; i < 4; ++i) {
    int d = t + i * 128;
    float ue = ur[d], ie = ir[d], ce = cr[d];
    float s1  = ue + cs[d];
    float ucm = 0.5f * (s1 * s1 - (ue * ue + css[d]));
    float s2 = ue + ie + ce + ucm;
    float q2 = ue * ue + ie * ie + ce * ce + ucm * ucm;
    float fm = 0.5f * (s2 * s2 - q2);
    fm_out[(size_t)b * DMODEL + d] = (bf16)fm;
    regl += ue * ue + ie * ie + ce * ce;
  }
  red[t] = regl;
  __syncthreads();
  #pragma unroll
  for (int s = 64; s > 0; s >>= 1) {
    if (t < s) red[t] += red[t + s];
    __syncthreads();
  }
  if (t == 0) atomicAdd(reg_acc, red[0]);
}

// ------------------------------------------------------- bf16 WMMA GEMM
// C[M,N] = relu(A[M,K] @ W[K,N] + bias), M=16384, N=K=512.
// WT is W transposed to [N][K] bf16 so each B fragment is two 16-byte contiguous runs.
// Block = 256 thr = 8 waves; block tile 256(M) x 64(N); wave tile 32 x 64.
// K-loop register double-buffered: loads for step k+1 issue before wmma of k.
__device__ __forceinline__ void load_a(Frag16& f, const bf16* Arow, int k0, int h) {
  // A fragment (16x32 bf16): lane(ln,h) elems 0..7 -> K=k0+8h+e,
  //                          elems 8..15 -> K=k0+16+8h+(e-8)
  f.q[0] = *(const uint4*)(Arow + k0 + 8 * h);
  f.q[1] = *(const uint4*)(Arow + k0 + 16 + 8 * h);
}
__device__ __forceinline__ void load_b(Frag16& f, const bf16* Bcol, int k0, int h) {
  // B fragment (32x16 bf16): lane(ln,h) elems 0..7 -> K=k0+8h+e,
  //                          elems 8..15 -> K=k0+16+8h+(e-8)
  f.q[0] = *(const uint4*)(Bcol + k0 + 8 * h);
  f.q[1] = *(const uint4*)(Bcol + k0 + 16 + 8 * h);
}

__global__ __launch_bounds__(256)
void k_gemm_bf16_wmma(const bf16* __restrict__ A, const bf16* __restrict__ WT,
                      const float* __restrict__ bias, bf16* __restrict__ Out) {
  int lane  = threadIdx.x & 31;
  int wave  = threadIdx.x >> 5;
  int mbase = blockIdx.x * 256 + wave * 32;
  int nbase = blockIdx.y * 64;
  int ln = lane & 15;     // row-within-tile (A) / col-within-tile (B,C)
  int h  = lane >> 4;     // lane-half selector per ISA fragment layout

  const bf16* Arow0 = A + (size_t)(mbase + ln) * DMODEL;
  const bf16* Arow1 = A + (size_t)(mbase + 16 + ln) * DMODEL;
  const bf16* Bc[4];
  #pragma unroll
  for (int nt = 0; nt < 4; ++nt)
    Bc[nt] = WT + (size_t)(nbase + nt * 16 + ln) * DMODEL;

  v8f acc[2][4];
  #pragma unroll
  for (int mt = 0; mt < 2; ++mt)
    #pragma unroll
    for (int nt = 0; nt < 4; ++nt) acc[mt][nt] = (v8f){};

  // prologue: load K-step 0
  Frag16 a0, a1, b[4];
  load_a(a0, Arow0, 0, h);
  load_a(a1, Arow1, 0, h);
  #pragma unroll
  for (int nt = 0; nt < 4; ++nt) load_b(b[nt], Bc[nt], 0, h);

  #pragma unroll
  for (int k0 = 0; k0 < DMODEL; k0 += 32) {
    // issue next K-step's loads before consuming this step's fragments
    Frag16 na0, na1, nb[4];
    if (k0 + 32 < DMODEL) {
      load_a(na0, Arow0, k0 + 32, h);
      load_a(na1, Arow1, k0 + 32, h);
      #pragma unroll
      for (int nt = 0; nt < 4; ++nt) load_b(nb[nt], Bc[nt], k0 + 32, h);
    }
    #pragma unroll
    for (int nt = 0; nt < 4; ++nt) {
      acc[0][nt] = __builtin_amdgcn_wmma_f32_16x16x32_bf16(
          false, a0.v, false, b[nt].v, (short)0, acc[0][nt], false, false);
      acc[1][nt] = __builtin_amdgcn_wmma_f32_16x16x32_bf16(
          false, a1.v, false, b[nt].v, (short)0, acc[1][nt], false, false);
    }
    a0 = na0; a1 = na1;
    #pragma unroll
    for (int nt = 0; nt < 4; ++nt) b[nt] = nb[nt];
  }

  // Epilogue: bias + ReLU + bf16 store.
  // C/D layout: vec elem r -> row = mtile_base + r + 8h, col = nbase + nt*16 + ln
  #pragma unroll
  for (int mt = 0; mt < 2; ++mt) {
    int mrow = mbase + mt * 16;
    #pragma unroll
    for (int nt = 0; nt < 4; ++nt) {
      int col  = nbase + nt * 16 + ln;
      float bv = bias[col];
      #pragma unroll
      for (int r = 0; r < 8; ++r) {
        float v = acc[mt][nt][r] + bv;
        v = v > 0.f ? v : 0.f;
        Out[(size_t)(mrow + r + 8 * h) * DMODEL + col] = (bf16)v;
      }
    }
  }
}

// ------------------------------------- logits = h2 @ w3 + b3, sigmoid, MSE sum
__global__ __launch_bounds__(256)
void k_loss(const bf16* __restrict__ H2, const float* __restrict__ w3,
            const float* __restrict__ b3, const float* __restrict__ rate,
            float* __restrict__ mse_acc) {
  int lane = threadIdx.x & 31;
  int wave = threadIdx.x >> 5;
  int row  = blockIdx.x * 8 + wave;

  const bf16* hr = H2 + (size_t)row * DMODEL + lane * 16;
  Frag16 hfr;
  hfr.q[0] = *(const uint4*)(hr);
  hfr.q[1] = *(const uint4*)(hr + 8);
  float s = 0.f;
  #pragma unroll
  for (int j = 0; j < 16; ++j)
    s += (float)hfr.e[j] * w3[lane * 16 + j];
  #pragma unroll
  for (int off = 16; off > 0; off >>= 1)
    s += __shfl_xor(s, off, 32);
  if (lane == 0) {
    float logit = s + b3[0];
    float pred  = 1.f + 4.f / (1.f + __expf(-logit));
    float diff  = pred - rate[row];
    atomicAdd(mse_acc, diff * diff);
  }
}

__global__ void k_finalize(const float* mse, const float* reg, float* out) {
  out[0] = mse[0] / (float)BATCH + 1e-4f * (0.5f * reg[0]) / (float)BATCH;
}

// ---------------------------------------------------------------------------
extern "C" void kernel_launch(void* const* d_in, const int* in_sizes, int n_in,
                              void* d_out, int out_size, void* d_ws, size_t ws_size,
                              hipStream_t stream) {
  (void)in_sizes; (void)n_in; (void)out_size; (void)ws_size;
  const int*   user   = (const int*)  d_in[0];
  const int*   item   = (const int*)  d_in[1];
  const int*   cate   = (const int*)  d_in[2];
  const float* rate   = (const float*)d_in[3];
  const float* user_w = (const float*)d_in[4];
  const float* item_w = (const float*)d_in[5];
  const float* cate_w = (const float*)d_in[6];
  const float* prior  = (const float*)d_in[7];
  const float* w1     = (const float*)d_in[8];
  const float* b1     = (const float*)d_in[9];
  const float* w2     = (const float*)d_in[10];
  const float* b2     = (const float*)d_in[11];
  const float* w3     = (const float*)d_in[12];
  const float* b3     = (const float*)d_in[13];
  float* out = (float*)d_out;

  // workspace layout (bytes)
  char* ws = (char*)d_ws;
  const size_t SZ_ACT = (size_t)BATCH * DMODEL * sizeof(bf16);   // 16 MB
  const size_t SZ_W   = (size_t)DMODEL * DMODEL * sizeof(bf16);  // 512 KB
  bf16*  fmA  = (bf16*)(ws);                         // fm input, reused for h2
  bf16*  h1   = (bf16*)(ws + SZ_ACT);
  bf16*  w1T  = (bf16*)(ws + 2 * SZ_ACT);
  bf16*  w2T  = (bf16*)(ws + 2 * SZ_ACT + SZ_W);
  float* cs   = (float*)(ws + 2 * SZ_ACT + 2 * SZ_W);
  float* css  = cs + DMODEL;
  float* regA = css + DMODEL;
  float* mseA = regA + 1;

  k_zero_init<<<1, DMODEL, 0, stream>>>(cs, css, regA, mseA);

  k_cate_stats<<<dim3(DMODEL / 256, NCATE / 200), 256, 0, stream>>>(cate_w, prior, cs, css);

  dim3 cg(DMODEL / 16, DMODEL / 16), cb(16, 16);
  k_convert_wT<<<cg, cb, 0, stream>>>(w1, w1T);
  k_convert_wT<<<cg, cb, 0, stream>>>(w2, w2T);

  k_fm<<<BATCH, 128, 0, stream>>>(user, item, cate, user_w, item_w, cate_w,
                                  cs, css, fmA, regA);

  dim3 gg(BATCH / 256, DMODEL / 64);
  k_gemm_bf16_wmma<<<gg, 256, 0, stream>>>(fmA, w1T, b1, h1);   // h1 = relu(fm@w1+b1)
  k_gemm_bf16_wmma<<<gg, 256, 0, stream>>>(h1, w2T, b2, fmA);   // h2 = relu(h1@w2+b2)

  k_loss<<<BATCH / 8, 256, 0, stream>>>(fmA, w3, b3, rate, mseA);

  k_finalize<<<1, 1, 0, stream>>>(mseA, regA, out);
}
